// Model_55748675502640
// MI455X (gfx1250) — matrix operand order repaired and checked by reading, not compiled, not run on hardware
//
#include <hip/hip_runtime.h>
#include <hip/hip_bf16.h>

// ---------------------------------------------------------------------------
// Shapes (compile-time constants from the reference)
// ---------------------------------------------------------------------------
#define Tt 512
#define Bb 256
#define Ss 64
#define Hh 128
#define Aa 8
#define TB (Tt * Bb)  // 131072

typedef __attribute__((ext_vector_type(16))) _Float16 v16h;
typedef __attribute__((ext_vector_type(8)))  float    v8f;

union Frag16 {           // 16 halfs = 8 VGPRs = one WMMA A or B operand
    v16h v;
    uint4 q[2];
};

__device__ __forceinline__ float sigmoid_f(float x) {
    return 1.0f / (1.0f + __expf(-x));
}
__device__ __forceinline__ float tanh_f(float x) {
    // overflow-safe tanh via exp
    float t = __expf(-2.0f * fabsf(x));
    float r = (1.0f - t) / (1.0f + t);
    return copysignf(r, x);
}

// ---------------------------------------------------------------------------
// Kernel 1: feat = relu(relu(state @ w1.T + b1) @ w2.T + b2)   -> f16 ws
//   grid.x = TB/128 blocks, 256 threads (8 waves), each wave = 16 rows.
// WMMA fragment conventions (16-bit, wave32):
//   A (16x32): lane l -> M=l&15, koA=(l>>4)*8; VGPR0-3 = K[ko..ko+7],
//              VGPR4-7 = K[16+ko..16+ko+7]   (two contiguous 16B chunks)
//   B (32x16): lane l -> N=l&15, kbB=(l>>4)*8; VGPR0-3 = K[kb..kb+7], VGPR4-7 = K[16+kb..16+kb+7]
//              (two contiguous 16B chunks of the weight row, since B[k][n]=W[n][k])
//   C/D (16x16 f32): VGPR r, lane l -> M = r + 8*(l>>4), N = l&15
// ---------------------------------------------------------------------------
__global__ void mlp_kernel(const float* __restrict__ state,  // [TB, 64]
                           const float* __restrict__ w1,     // [128, 64]
                           const float* __restrict__ b1,     // [128]
                           const float* __restrict__ w2,     // [128, 128]
                           const float* __restrict__ b2,     // [128]
                           _Float16* __restrict__ feat)      // [TB, 128] f16
{
    extern __shared__ char smem_raw[];
    _Float16* w1s = (_Float16*)smem_raw;           // 128*64
    _Float16* w2s = w1s + 128 * 64;                // 128*128
    _Float16* xs  = w2s + 128 * 128;               // 8 waves * 16*64
    _Float16* h1s = xs + 8 * 16 * 64;              // 8 waves * 16*128
    float*    b1s = (float*)(h1s + 8 * 16 * 128);  // 128
    float*    b2s = b1s + 128;                     // 128

    const int tid = threadIdx.x;
    for (int i = tid; i < 128 * 64; i += 256)  w1s[i] = (_Float16)w1[i];
    for (int i = tid; i < 128 * 128; i += 256) w2s[i] = (_Float16)w2[i];
    if (tid < 128) { b1s[tid] = b1[tid]; b2s[tid] = b2[tid]; }
    __syncthreads();

    const int wave = tid >> 5, lane = tid & 31;
    const int M    = lane & 15;
    const int koA  = (lane >> 4) * 8;
    const int kbB  = (lane >> 4) * 8;
    const int nl   = lane & 15;
    const int rbase = (lane >> 4) * 8;
    const int rowbase = (blockIdx.x * 8 + wave) * 16;

    _Float16* xw = xs + wave * 16 * 64;
    _Float16* hw = h1s + wave * 16 * 128;

    // stage 16x64 state tile as f16 (wave-private region)
    for (int i = lane; i < 16 * 64; i += 32) {
        int r = i >> 6, c = i & 63;
        xw[i] = (_Float16)state[(size_t)(rowbase + r) * Ss + c];
    }
    __syncthreads();

    // ---- GEMM1: 16x64 @ 64x128 -> h1 ----
    v8f acc[8];
#pragma unroll
    for (int nt = 0; nt < 8; nt++) {
        float bv = b1s[nt * 16 + nl];
#pragma unroll
        for (int e = 0; e < 8; e++) acc[nt][e] = bv;
    }
#pragma unroll
    for (int kt = 0; kt < 2; kt++) {
        Frag16 af;
        const _Float16* ap = xw + M * 64 + kt * 32 + koA;
        af.q[0] = *(const uint4*)(ap);
        af.q[1] = *(const uint4*)(ap + 16);
#pragma unroll
        for (int nt = 0; nt < 8; nt++) {
            Frag16 bf;
            const _Float16* wp = w1s + (nt * 16 + nl) * 64 + kt * 32 + kbB;
            bf.q[0] = ((const uint4*)wp)[0];
            bf.q[1] = ((const uint4*)wp)[2];
            acc[nt] = __builtin_amdgcn_wmma_f32_16x16x32_f16(
                false, af.v, false, bf.v, (short)0, acc[nt], false, false);
        }
    }
#pragma unroll
    for (int nt = 0; nt < 8; nt++) {
#pragma unroll
        for (int r = 0; r < 8; r++) {
            float v = acc[nt][r];
            v = v > 0.0f ? v : 0.0f;
            hw[(rbase + r) * 128 + nt * 16 + nl] = (_Float16)v;
        }
    }
    __syncthreads();

    // ---- GEMM2: 16x128 @ 128x128 -> feat ----
#pragma unroll
    for (int nt = 0; nt < 8; nt++) {
        float bv = b2s[nt * 16 + nl];
#pragma unroll
        for (int e = 0; e < 8; e++) acc[nt][e] = bv;
    }
#pragma unroll
    for (int kt = 0; kt < 4; kt++) {
        Frag16 af;
        const _Float16* ap = hw + M * 128 + kt * 32 + koA;
        af.q[0] = *(const uint4*)(ap);
        af.q[1] = *(const uint4*)(ap + 16);
#pragma unroll
        for (int nt = 0; nt < 8; nt++) {
            Frag16 bf;
            const _Float16* wp = w2s + (nt * 16 + nl) * 128 + kt * 32 + kbB;
            bf.q[0] = ((const uint4*)wp)[0];
            bf.q[1] = ((const uint4*)wp)[2];
            acc[nt] = __builtin_amdgcn_wmma_f32_16x16x32_f16(
                false, af.v, false, bf.v, (short)0, acc[nt], false, false);
        }
    }
#pragma unroll
    for (int nt = 0; nt < 8; nt++) {
#pragma unroll
        for (int r = 0; r < 8; r++) {
            float v = acc[nt][r];
            v = v > 0.0f ? v : 0.0f;
            feat[(size_t)(rowbase + rbase + r) * Hh + nt * 16 + nl] = (_Float16)v;
        }
    }
}

// ---------------------------------------------------------------------------
// Kernel 2: fused gate-projection + LSTM scan + output head.
//   grid.x = 16 (batch tiles of 16 rows), 256 threads (8 waves).
//   Wave w owns hidden columns [16w,16w+16) across all 4 gates.
//   Per step: gates = bias + feat[t]@wih.T + h@whh.T  (32 WMMAs/wave),
//   elementwise LSTM cell (c kept in the f32 accumulator register layout),
//   h double-buffered in LDS (carry = masked, out = pre-mask), then wave 0
//   runs the head GEMM (mean|logstd stacked, or value) on the pre-mask h.
//   mode 0: actor (out0=mean, out1=std);  mode 1: critic (out0=value)
// ---------------------------------------------------------------------------
__global__ void lstm_kernel(const _Float16* __restrict__ feat,  // [TB,128] f16
                            const float* __restrict__ wih,      // [512,128]
                            const float* __restrict__ whh,      // [512,128]
                            const float* __restrict__ bih,      // [512]
                            const float* __restrict__ bhh,      // [512]
                            const int*   __restrict__ dones,    // [TB]
                            const float* __restrict__ hw0,      // mean_w / val_w
                            const float* __restrict__ hw1,      // logstd_w / null
                            const float* __restrict__ hb0,      // mean_b / val_b
                            const float* __restrict__ hb1,      // logstd_b / null
                            float* __restrict__ out0,
                            float* __restrict__ out1,
                            int mode)
{
    extern __shared__ char smem_raw[];
    _Float16* wih_s  = (_Float16*)smem_raw;          // 512*128
    _Float16* whh_s  = wih_s + 512 * 128;            // 512*128
    _Float16* headw  = whh_s + 512 * 128;            // 16*128
    _Float16* hcar   = headw + 16 * 128;             // 2 * 16*128
    _Float16* hout   = hcar + 2 * 16 * 128;          // 2 * 16*128
    float*    bias_s = (float*)(hout + 2 * 16 * 128);// 512
    float*    hbias  = bias_s + 512;                 // 16

    const int tid = threadIdx.x;
    for (int i = tid; i < 512 * 128; i += 256) {
        wih_s[i] = (_Float16)wih[i];
        whh_s[i] = (_Float16)whh[i];
    }
    for (int i = tid; i < 512; i += 256) bias_s[i] = bih[i] + bhh[i];
    for (int i = tid; i < 16 * 128; i += 256) {
        int rr = i >> 7, cc = i & 127;
        float v = 0.0f;
        if (mode == 0) v = (rr < 8) ? hw0[rr * 128 + cc] : hw1[(rr - 8) * 128 + cc];
        else if (rr == 0) v = hw0[cc];
        headw[i] = (_Float16)v;
    }
    if (tid < 16) {
        float v = 0.0f;
        if (mode == 0) v = (tid < 8) ? hb0[tid] : hb1[tid - 8];
        else if (tid == 0) v = hb0[0];
        hbias[tid] = v;
    }
    for (int i = tid; i < 2 * 16 * 128; i += 256) hcar[i] = (_Float16)0.0f;
    __syncthreads();

    const int wave = tid >> 5, lane = tid & 31;
    const int M    = lane & 15;
    const int koA  = (lane >> 4) * 8;
    const int kbB  = (lane >> 4) * 8;
    const int nl   = lane & 15;
    const int rbase = (lane >> 4) * 8;
    const int bbase = blockIdx.x * 16;
    const int col   = wave * 16 + nl;

    float bg[4];
#pragma unroll
    for (int g = 0; g < 4; g++) bg[g] = bias_s[g * 128 + col];
    const float hb = hbias[nl];

    v8f c;
#pragma unroll
    for (int e = 0; e < 8; e++) c[e] = 0.0f;

    for (int t = 0; t < Tt; t++) {
        const int par = t & 1;
        v8f acc[4];
#pragma unroll
        for (int g = 0; g < 4; g++)
#pragma unroll
            for (int e = 0; e < 8; e++) acc[g][e] = bg[g];

        // ---- feat[t] @ wih.T  (K = 0..127) ----
        const _Float16* frow = feat + ((size_t)t * Bb + bbase + M) * Hh;
#pragma unroll
        for (int kt = 0; kt < 4; kt++) {
            Frag16 af;
            af.q[0] = *(const uint4*)(frow + kt * 32 + koA);
            af.q[1] = *(const uint4*)(frow + kt * 32 + 16 + koA);
#pragma unroll
            for (int g = 0; g < 4; g++) {
                Frag16 bf;
                const _Float16* wp = wih_s + (g * 128 + col) * 128 + kt * 32 + kbB;
                bf.q[0] = ((const uint4*)wp)[0];
                bf.q[1] = ((const uint4*)wp)[2];
                acc[g] = __builtin_amdgcn_wmma_f32_16x16x32_f16(
                    false, af.v, false, bf.v, (short)0, acc[g], false, false);
            }
        }
        // ---- h @ whh.T  (K = 0..127) ----
        const _Float16* hrow = hcar + par * (16 * 128) + M * 128;
#pragma unroll
        for (int kt = 0; kt < 4; kt++) {
            Frag16 af;
            af.q[0] = *(const uint4*)(hrow + kt * 32 + koA);
            af.q[1] = *(const uint4*)(hrow + kt * 32 + 16 + koA);
#pragma unroll
            for (int g = 0; g < 4; g++) {
                Frag16 bf;
                const _Float16* wp = whh_s + (g * 128 + col) * 128 + kt * 32 + kbB;
                bf.q[0] = ((const uint4*)wp)[0];
                bf.q[1] = ((const uint4*)wp)[2];
                acc[g] = __builtin_amdgcn_wmma_f32_16x16x32_f16(
                    false, af.v, false, bf.v, (short)0, acc[g], false, false);
            }
        }

        // ---- LSTM cell (per-lane 8 elements, rows rbase..rbase+7, col) ----
        const int* dp = dones + (size_t)t * Bb + bbase + rbase;
#pragma unroll
        for (int r = 0; r < 8; r++) {
            float iv = sigmoid_f(acc[0][r]);
            float fv = sigmoid_f(acc[1][r]);
            float gv = tanh_f(acc[2][r]);
            float ov = sigmoid_f(acc[3][r]);
            float c2 = fv * c[r] + iv * gv;
            float h2 = ov * tanh_f(c2);
            float m  = 1.0f - (float)dp[r];
            c[r] = c2 * m;
            int row = rbase + r;
            hout[par * (16 * 128) + row * 128 + col]        = (_Float16)h2;
            hcar[(par ^ 1) * (16 * 128) + row * 128 + col]  = (_Float16)(h2 * m);
        }
        __syncthreads();

        // ---- head: one 16x16 tile from pre-mask h (wave 0 only) ----
        if (wave == 0) {
            v8f hd;
#pragma unroll
            for (int e = 0; e < 8; e++) hd[e] = hb;
            const _Float16* ho = hout + par * (16 * 128) + M * 128;
#pragma unroll
            for (int kt = 0; kt < 4; kt++) {
                Frag16 af;
                af.q[0] = *(const uint4*)(ho + kt * 32 + koA);
                af.q[1] = *(const uint4*)(ho + kt * 32 + 16 + koA);
                Frag16 bf;
                const _Float16* wp = headw + nl * 128 + kt * 32 + kbB;
                bf.q[0] = ((const uint4*)wp)[0];
                bf.q[1] = ((const uint4*)wp)[2];
                hd = __builtin_amdgcn_wmma_f32_16x16x32_f16(
                    false, af.v, false, bf.v, (short)0, hd, false, false);
            }
#pragma unroll
            for (int r = 0; r < 8; r++) {
                size_t idx = (size_t)t * Bb + bbase + rbase + r;
                if (mode == 0) {
                    if (nl < 8) out0[idx * Aa + nl] = hd[r];
                    else        out1[idx * Aa + (nl - 8)] = __expf(hd[r]);
                } else {
                    if (nl == 0) out0[idx] = hd[r];
                }
            }
        }
        __syncthreads();
    }
}

// ---------------------------------------------------------------------------
// Host launcher
// ---------------------------------------------------------------------------
extern "C" void kernel_launch(void* const* d_in, const int* in_sizes, int n_in,
                              void* d_out, int out_size, void* d_ws, size_t ws_size,
                              hipStream_t stream) {
    const float* state  = (const float*)d_in[0];
    const int*   dones  = (const int*)d_in[1];
    const float* a_w1   = (const float*)d_in[2];
    const float* a_b1   = (const float*)d_in[3];
    const float* a_w2   = (const float*)d_in[4];
    const float* a_b2   = (const float*)d_in[5];
    const float* c_w1   = (const float*)d_in[6];
    const float* c_b1   = (const float*)d_in[7];
    const float* c_w2   = (const float*)d_in[8];
    const float* c_b2   = (const float*)d_in[9];
    const float* a_wih  = (const float*)d_in[10];
    const float* a_whh  = (const float*)d_in[11];
    const float* a_bih  = (const float*)d_in[12];
    const float* a_bhh  = (const float*)d_in[13];
    const float* c_wih  = (const float*)d_in[14];
    const float* c_whh  = (const float*)d_in[15];
    const float* c_bih  = (const float*)d_in[16];
    const float* c_bhh  = (const float*)d_in[17];
    const float* mean_w = (const float*)d_in[18];
    const float* mean_b = (const float*)d_in[19];
    const float* lstd_w = (const float*)d_in[20];
    const float* lstd_b = (const float*)d_in[21];
    const float* val_w  = (const float*)d_in[22];
    const float* val_b  = (const float*)d_in[23];

    float* mean_o = (float*)d_out;
    float* std_o  = mean_o + (size_t)TB * Aa;
    float* val_o  = std_o + (size_t)TB * Aa;

    _Float16* featA = (_Float16*)d_ws;
    _Float16* featC = featA + (size_t)TB * Hh;

    const size_t mlp_smem  = (size_t)(128 * 64 + 128 * 128 + 8 * 16 * 64 + 8 * 16 * 128) * 2
                           + 2 * 128 * 4;
    const size_t lstm_smem = (size_t)(512 * 128 * 2 + 16 * 128 + 4 * 16 * 128) * 2
                           + (512 + 16) * 4;

    (void)hipFuncSetAttribute((const void*)mlp_kernel,
                              hipFuncAttributeMaxDynamicSharedMemorySize, (int)mlp_smem);
    (void)hipFuncSetAttribute((const void*)lstm_kernel,
                              hipFuncAttributeMaxDynamicSharedMemorySize, (int)lstm_smem);

    dim3 mgrid(TB / 128);
    mlp_kernel<<<mgrid, 256, mlp_smem, stream>>>(state, a_w1, a_b1, a_w2, a_b2, featA);
    mlp_kernel<<<mgrid, 256, mlp_smem, stream>>>(state, c_w1, c_b1, c_w2, c_b2, featC);

    dim3 lgrid(Bb / 16);
    lstm_kernel<<<lgrid, 256, lstm_smem, stream>>>(
        featA, a_wih, a_whh, a_bih, a_bhh, dones,
        mean_w, lstd_w, mean_b, lstd_b, mean_o, std_o, 0);
    lstm_kernel<<<lgrid, 256, lstm_smem, stream>>>(
        featC, c_wih, c_whh, c_bih, c_bhh, dones,
        val_w, (const float*)nullptr, val_b, (const float*)nullptr, val_o, (float*)nullptr, 1);
}
